// PieperCausalIK_6665789243796
// MI455X (gfx1250) — matrix operand order repaired and checked by reading, not compiled, not run on hardware
//
#include <hip/hip_runtime.h>
#include <hip/hip_bf16.h>
#include <math.h>

// ---------------------------------------------------------------------------
// PieperCausalIK on MI455X (gfx1250, wave32, WMMA).
// All heavy GEMMs use v_wmma_f32_16x16x32_f16 (f16 A/B, f32 accumulate).
// ---------------------------------------------------------------------------

typedef __attribute__((ext_vector_type(16))) _Float16 v16h;
typedef __attribute__((ext_vector_type(8)))  _Float16 v8h;
typedef __attribute__((ext_vector_type(8)))  float    v8f;

static constexpr int Bn = 32768;   // batch
static constexpr int Hn = 256;     // hidden
static constexpr int Tn = 10;      // frames

// ---- WMMA fragment helpers (layouts per CDNA5 ISA 7.12.2) ------------------
// A (16x32 f16): lane<16 -> row=lane, K = f*32 + {0..7, 16..23}
//                lane>=16 -> row=lane-16, K = f*32 + {8..15, 24..31}
__device__ inline v16h load_a_frag(const _Float16* rowbase, int kbase, int hi8) {
  v8h lo = *(const v8h*)(rowbase + kbase + hi8);
  v8h hh = *(const v8h*)(rowbase + kbase + 16 + hi8);
  return __builtin_shufflevector(lo, hh, 0,1,2,3,4,5,6,7,8,9,10,11,12,13,14,15);
}
// B (32x16 f16): lane<16 -> col=lane, K = f*32 + {0..7, 16..23}
//                lane>=16 -> col=lane-16, K = f*32 + {8..15, 24..31}
__device__ inline v16h load_b_frag(const _Float16* colbase, int kbase, int hi16) {
  v8h lo = *(const v8h*)(colbase + kbase + hi16 / 2);
  v8h hh = *(const v8h*)(colbase + kbase + 16 + hi16 / 2);
  return __builtin_shufflevector(lo, hh, 0,1,2,3,4,5,6,7,8,9,10,11,12,13,14,15);
}
__device__ inline v8f wmma16(v16h a, v16h b, v8f c) {
  return __builtin_amdgcn_wmma_f32_16x16x32_f16(false, a, false, b, (short)0, c,
                                                false, false);
}

// ---- Weight prep: f32 [K,N] -> f16 [N,K] (transposed, contiguous K) --------
__global__ __launch_bounds__(256) void transpose_f16_k(
    const float* __restrict__ w, _Float16* __restrict__ wt, int K, int N) {
  int idx = blockIdx.x * 256 + threadIdx.x;
  if (idx >= K * N) return;
  int n = idx % N, k = idx / N;
  wt[(size_t)n * K + k] = (_Float16)w[(size_t)k * N + n];
}

// conv w2 [O=256, I=256, kt=3] f32 -> wt[(kt*256+o)*256+c] f16
__global__ __launch_bounds__(256) void prep_conv_k(
    const float* __restrict__ w, _Float16* __restrict__ wt) {
  int idx = blockIdx.x * 256 + threadIdx.x;
  if (idx >= 3 * 256 * 256) return;
  int c = idx & 255, o = (idx >> 8) & 255, kt = idx >> 16;
  wt[idx] = (_Float16)w[(o * 256 + c) * 3 + kt];
}

// ---- Fused temporal block: conv1(relu) -> conv2 (WMMA, relu) -> mean ------
// One block = 16 batch rows. h1 tile kept in LDS (80 KB).
__global__ __launch_bounds__(256) void temporal_fused(
    const float* __restrict__ hist,    // [B,10,7]
    const float* __restrict__ w1,      // [256, cin, 3]
    const float* __restrict__ b1,      // [256]
    const _Float16* __restrict__ w2t,  // [3][256][256]  (kt, out, in)
    const float* __restrict__ b2,      // [256]
    _Float16* __restrict__ feat,       // [B,256]
    int ch0, int cin) {
  __shared__ __align__(16) _Float16 h1[Tn * 16 * 256];  // 80 KB
  __shared__ float xl[16 * Tn * 3];
  const int tid = threadIdx.x;
  const int r0 = blockIdx.x * 16;

  // stage input slice [16 rows][10 t][cin]
  for (int e = tid; e < 16 * Tn * cin; e += 256) {
    int c = e % cin, t = (e / cin) % Tn, r = e / (cin * Tn);
    xl[(r * Tn + t) * 3 + c] = hist[(size_t)(r0 + r) * 70 + t * 7 + ch0 + c];
  }
  __syncthreads();

  // conv1 + relu  (tiny K: VALU)
  for (int e = tid; e < Tn * 16 * 256; e += 256) {
    int o = e & 255, r = (e >> 8) & 15, t = e >> 12;
    float acc = b1[o];
    for (int kt = 0; kt < 3; ++kt) {
      int tt = t + kt - 1;
      if (tt < 0 || tt >= Tn) continue;
      for (int c = 0; c < cin; ++c)
        acc += xl[(r * Tn + tt) * 3 + c] * w1[(o * cin + c) * 3 + kt];
    }
    h1[(t * 16 + r) * 256 + o] = (_Float16)fmaxf(acc, 0.f);
  }
  __syncthreads();

  // conv2 via WMMA (K = 256 per conv tap), relu, mean over T
  const int lane = tid & 31, wave = tid >> 5;
  const int mrow = lane & 15, hi = lane >> 4, hi8 = hi * 8, hi16 = hi * 16;
  for (int nt = wave * 2; nt < wave * 2 + 2; ++nt) {
    const int col = nt * 16 + mrow;
    const float bv = b2[col];
    v8f facc = {0, 0, 0, 0, 0, 0, 0, 0};
    for (int t = 0; t < Tn; ++t) {
      v8f acc = {bv, bv, bv, bv, bv, bv, bv, bv};
      for (int kt = 0; kt < 3; ++kt) {
        int tt = t + kt - 1;
        if (tt < 0 || tt >= Tn) continue;   // uniform across wave
        const _Float16* arow = &h1[(tt * 16 + mrow) * 256];
        const _Float16* cbase = w2t + ((size_t)kt * 256 + col) * 256;
#pragma unroll
        for (int f = 0; f < 8; ++f) {
          v16h a = load_a_frag(arow, f * 32, hi8);
          v16h b = load_b_frag(cbase, f * 32, hi16);
          acc = wmma16(a, b, acc);
        }
      }
#pragma unroll
      for (int i = 0; i < 8; ++i) facc[i] += fmaxf(acc[i], 0.f) * 0.1f;
    }
#pragma unroll
    for (int i = 0; i < 8; ++i)
      feat[(size_t)(r0 + i + hi8) * 256 + col] = (_Float16)facc[i];
  }
}

// ---- Generic WMMA GEMM: io = epilogue(A1|A2 @ Wt^T + bias) ----------------
// M = 32768, N = 256 fixed. A-frags register resident; 8 waves/block.
// MODE 0: io = y ; MODE 1: io += y ; MODE 2: g=sigmoid(y), io = io*(1-g)+y*g
template <int NF1, int NF2, int MODE>
__global__ __launch_bounds__(256) void gemm_f16(
    const _Float16* __restrict__ A1, const _Float16* __restrict__ A2,
    const _Float16* __restrict__ Wt, const float* __restrict__ bias,
    _Float16* __restrict__ io) {
  constexpr int NF = NF1 + NF2;
  constexpr int K1 = NF1 * 32, K2 = NF2 * 32, KT = NF * 32;
  const int lane = threadIdx.x & 31, wave = threadIdx.x >> 5;
  const int m0 = (blockIdx.x * 8 + wave) * 16;
  const int mrow = lane & 15, hi = lane >> 4, hi8 = hi * 8, hi16 = hi * 16;

  v16h af[NF];
  {
    const _Float16* r1 = A1 + (size_t)(m0 + mrow) * K1;
#pragma unroll
    for (int f = 0; f < NF1; ++f) af[f] = load_a_frag(r1, f * 32, hi8);
  }
  if constexpr (NF2 > 0) {
    const _Float16* r2 = A2 + (size_t)(m0 + mrow) * K2;
#pragma unroll
    for (int f = 0; f < NF2; ++f) af[NF1 + f] = load_a_frag(r2, f * 32, hi8);
  }
  for (int nt = 0; nt < 16; ++nt) {
    const int col = nt * 16 + mrow;
    const float bv = bias[col];
    v8f acc = {bv, bv, bv, bv, bv, bv, bv, bv};
    const _Float16* cb = Wt + (size_t)col * KT;
#pragma unroll
    for (int f = 0; f < NF; ++f) {
      v16h bf = load_b_frag(cb, f * 32, hi16);
      acc = wmma16(af[f], bf, acc);
    }
#pragma unroll
    for (int i = 0; i < 8; ++i) {
      const size_t idx = (size_t)(m0 + i + hi8) * 256 + col;
      float y = acc[i];
      if (MODE == 0) {
        io[idx] = (_Float16)y;
      } else if (MODE == 1) {
        io[idx] = (_Float16)(y + (float)io[idx]);
      } else {
        float g = 1.f / (1.f + expf(-y));
        float prev = (float)io[idx];
        io[idx] = (_Float16)(prev * (1.f - g) + y * g);
      }
    }
  }
}

// ---- Small first layers (K = 3 or 4) --------------------------------------
__global__ __launch_bounds__(256) void lin_small_k(
    const float* __restrict__ X, const float* __restrict__ W,
    const float* __restrict__ bias, _Float16* __restrict__ out, int Din) {
  int idx = blockIdx.x * 256 + threadIdx.x;  // b*128 + j
  int j = idx & 127, bb = idx >> 7;
  float acc = bias[j];
  for (int d = 0; d < Din; ++d) acc += X[(size_t)bb * Din + d] * W[d * 128 + j];
  out[idx] = (_Float16)acc;
}

// ---- LayerNorm + exact GELU (one wave per row, wave32 shfl reduce) --------
__global__ __launch_bounds__(256) void ln_gelu_k(
    const _Float16* __restrict__ in, const float* __restrict__ g,
    const float* __restrict__ b, _Float16* __restrict__ out, int D) {
  const int lane = threadIdx.x & 31;
  const int row = blockIdx.x * 8 + (threadIdx.x >> 5);
  const _Float16* x = in + (size_t)row * D;
  float s = 0.f, s2 = 0.f;
  for (int i = lane; i < D; i += 32) {
    float v = (float)x[i];
    s += v; s2 += v * v;
  }
#pragma unroll
  for (int off = 16; off; off >>= 1) {
    s  += __shfl_xor(s, off, 32);
    s2 += __shfl_xor(s2, off, 32);
  }
  float m = s / D;
  float inv = rsqrtf(s2 / D - m * m + 1e-5f);
  for (int i = lane; i < D; i += 32) {
    float v = ((float)x[i] - m) * inv * g[i] + b[i];
    float ge = 0.5f * v * (1.f + erff(v * 0.70710678118654752f));
    out[(size_t)row * D + i] = (_Float16)ge;
  }
}

// ---- FiLM combine: node = gamma * feat + beta (feat buffer in-place) ------
__global__ __launch_bounds__(256) void film_k(
    const _Float16* __restrict__ ga, const _Float16* __restrict__ be,
    _Float16* __restrict__ node) {
  size_t i = (size_t)blockIdx.x * 256 + threadIdx.x;
  node[i] = (_Float16)((float)ga[i] * (float)node[i] + (float)be[i]);
}

// ---- Heads: out[B,7] f32 ---------------------------------------------------
__global__ __launch_bounds__(256) void heads_k(
    const _Float16* __restrict__ ns, const _Float16* __restrict__ ne,
    const _Float16* __restrict__ nf, const _Float16* __restrict__ nw,
    const float* __restrict__ ws, const float* __restrict__ bs,
    const float* __restrict__ we, const float* __restrict__ be,
    const float* __restrict__ wf, const float* __restrict__ bf,
    const float* __restrict__ ww, const float* __restrict__ bw,
    float* __restrict__ out) {
  int idx = blockIdx.x * 256 + threadIdx.x;
  if (idx >= Bn * 7) return;
  int o = idx % 7, bb = idx / 7;
  const _Float16* node; const float* w; const float* bias; int col, nc;
  if (o < 3)      { node = ns; w = ws; bias = bs; col = o;     nc = 3; }
  else if (o == 3){ node = ne; w = we; bias = be; col = 0;     nc = 1; }
  else if (o == 4){ node = nf; w = wf; bias = bf; col = 0;     nc = 1; }
  else            { node = nw; w = ww; bias = bw; col = o - 5; nc = 2; }
  float acc = bias[col];
  for (int k = 0; k < 256; ++k)
    acc += (float)node[(size_t)bb * 256 + k] * w[k * nc + col];
  out[idx] = acc;
}

// ---------------------------------------------------------------------------
// Host launch. Input leaf order (setup_inputs dict insertion order, recursive):
//  0 history[B,10,7]  1 end_pos[B,3]  2 end_ori[B,4]
//  3..8   pos_enc  {w1[3,128], b1, ln_g, ln_b, w2[128,256], b2}
//  9..14  ori_enc  {w1[4,128], ...}
//  15..30 temporal shoulder/elbow/forearm/wrist {w1[256,cin,3], b1, w2[256,256,3], b2}
//  31..78 film per joint {gamma mlp 6 leaves, beta mlp 6 leaves}
//  79..90 msg layer0/1 {s2e_w, s2e_b, tf_w[512,256], tf_b, f2w_w[512,256], f2w_b}
//  91..98 heads per joint {w[256,oj], b}
// ---------------------------------------------------------------------------
extern "C" void kernel_launch(void* const* d_in, const int* in_sizes, int n_in,
                              void* d_out, int out_size, void* d_ws,
                              size_t ws_size, hipStream_t stream) {
  (void)in_sizes; (void)n_in; (void)out_size; (void)ws_size;
  auto F = [&](int i) { return (const float*)d_in[i]; };
  char* W = (char*)d_ws;
  const size_t ACT = (size_t)Bn * 256 * 2;  // 16 MB f16 activation buffer

  _Float16* feat[4];
  for (int j = 0; j < 4; ++j) feat[j] = (_Float16*)(W + (size_t)j * ACT);
  _Float16* target = (_Float16*)(W + 4 * ACT);
  _Float16* tmp1   = (_Float16*)(W + 5 * ACT);
  _Float16* tmp2   = (_Float16*)(W + 6 * ACT);
  _Float16* gammaB = (_Float16*)(W + 7 * ACT);
  char* wb = W + 8 * ACT;                       // f16 weight region
  _Float16* posW2t = (_Float16*)(wb);
  _Float16* oriW2t = (_Float16*)(wb + 65536);
  auto filmT = [&](int j, int gb, int m) {      // 16 x [256,256]
    return (_Float16*)(wb + 131072 + (size_t)(((j * 2 + gb) * 2 + m)) * 131072);
  };
  char* mb = wb + 131072 + 16 * 131072;
  auto s2eT = [&](int l) { return (_Float16*)(mb + (size_t)l * 655360); };
  auto tfT  = [&](int l) { return (_Float16*)(mb + (size_t)l * 655360 + 131072); };
  auto f2wT = [&](int l) { return (_Float16*)(mb + (size_t)l * 655360 + 393216); };
  char* cb = mb + 2 * 655360;
  auto convT = [&](int j) { return (_Float16*)(cb + (size_t)j * 393216); };

  // ---- weight prep --------------------------------------------------------
  transpose_f16_k<<<128, 256, 0, stream>>>(F(7),  posW2t, 128, 256);
  transpose_f16_k<<<128, 256, 0, stream>>>(F(13), oriW2t, 128, 256);
  for (int j = 0; j < 4; ++j)
    for (int gb = 0; gb < 2; ++gb) {
      int base = 31 + j * 12 + gb * 6;
      transpose_f16_k<<<256, 256, 0, stream>>>(F(base + 0), filmT(j, gb, 0), 256, 256);
      transpose_f16_k<<<256, 256, 0, stream>>>(F(base + 4), filmT(j, gb, 1), 256, 256);
    }
  for (int l = 0; l < 2; ++l) {
    int base = 79 + l * 6;
    transpose_f16_k<<<256, 256, 0, stream>>>(F(base + 0), s2eT(l), 256, 256);
    transpose_f16_k<<<512, 256, 0, stream>>>(F(base + 2), tfT(l),  512, 256);
    transpose_f16_k<<<512, 256, 0, stream>>>(F(base + 4), f2wT(l), 512, 256);
  }
  for (int j = 0; j < 4; ++j)
    prep_conv_k<<<768, 256, 0, stream>>>(F(15 + j * 4 + 2), convT(j));

  // ---- temporal feature extraction (dominant GEMM work) -------------------
  const int ch0[4] = {0, 3, 4, 5};
  const int cin[4] = {3, 1, 1, 2};
  for (int j = 0; j < 4; ++j) {
    int tb = 15 + j * 4;
    temporal_fused<<<Bn / 16, 256, 0, stream>>>(
        F(0), F(tb), F(tb + 1), convT(j), F(tb + 3), feat[j], ch0[j], cin[j]);
  }

  // ---- target = mlp(pos) + mlp(ori) ---------------------------------------
  lin_small_k<<<Bn * 128 / 256, 256, 0, stream>>>(F(1), F(3), F(4), tmp1, 3);
  ln_gelu_k<<<Bn / 8, 256, 0, stream>>>(tmp1, F(5), F(6), tmp2, 128);
  gemm_f16<4, 0, 0><<<Bn / 128, 256, 0, stream>>>(tmp2, nullptr, posW2t, F(8), target);
  lin_small_k<<<Bn * 128 / 256, 256, 0, stream>>>(F(2), F(9), F(10), tmp1, 4);
  ln_gelu_k<<<Bn / 8, 256, 0, stream>>>(tmp1, F(11), F(12), tmp2, 128);
  gemm_f16<4, 0, 1><<<Bn / 128, 256, 0, stream>>>(tmp2, nullptr, oriW2t, F(14), target);

  // ---- FiLM per joint: node_j = gamma*feat_j + beta -----------------------
  for (int j = 0; j < 4; ++j) {
    int bg = 31 + j * 12, bb = bg + 6;
    // gamma = MLP(target)
    gemm_f16<8, 0, 0><<<Bn / 128, 256, 0, stream>>>(target, nullptr, filmT(j, 0, 0), F(bg + 1), tmp1);
    ln_gelu_k<<<Bn / 8, 256, 0, stream>>>(tmp1, F(bg + 2), F(bg + 3), tmp2, 256);
    gemm_f16<8, 0, 0><<<Bn / 128, 256, 0, stream>>>(tmp2, nullptr, filmT(j, 0, 1), F(bg + 5), gammaB);
    // beta = MLP(target)   (lands in tmp1)
    gemm_f16<8, 0, 0><<<Bn / 128, 256, 0, stream>>>(target, nullptr, filmT(j, 1, 0), F(bb + 1), tmp1);
    ln_gelu_k<<<Bn / 8, 256, 0, stream>>>(tmp1, F(bb + 2), F(bb + 3), tmp2, 256);
    gemm_f16<8, 0, 0><<<Bn / 128, 256, 0, stream>>>(tmp2, nullptr, filmT(j, 1, 1), F(bb + 5), tmp1);
    film_k<<<Bn, 256, 0, stream>>>(gammaB, tmp1, feat[j]);  // feat[j] becomes node_j
  }

  // ---- causal chain message passing ---------------------------------------
  for (int l = 0; l < 2; ++l) {
    int base = 79 + l * 6;
    // elbow += shoulder @ s2e
    gemm_f16<8, 0, 1><<<Bn / 128, 256, 0, stream>>>(feat[0], nullptr, s2eT(l), F(base + 1), feat[1]);
    // forearm = gated(concat(shoulder, elbow) @ tf)
    gemm_f16<8, 8, 2><<<Bn / 128, 256, 0, stream>>>(feat[0], feat[1], tfT(l), F(base + 3), feat[2]);
    // wrist = gated(concat(elbow, forearm) @ f2w)
    gemm_f16<8, 8, 2><<<Bn / 128, 256, 0, stream>>>(feat[1], feat[2], f2wT(l), F(base + 5), feat[3]);
  }

  // ---- heads --------------------------------------------------------------
  heads_k<<<(Bn * 7 + 255) / 256, 256, 0, stream>>>(
      feat[0], feat[1], feat[2], feat[3],
      F(91), F(92), F(93), F(94), F(95), F(96), F(97), F(98), (float*)d_out);
}
